// FaST_65644280152707
// MI455X (gfx1250) — compile-verified, matrix operand order checked by reading, not run on hardware
//
#include <hip/hip_runtime.h>
#include <hip/hip_bf16.h>
#include <math.h>

// ---------------------------------------------------------------------------
// FaST forward pass for MI455X (gfx1250, wave32, WMMA bf16 16x16x32)
// B operands pre-packed into fragment-native layout -> b128 loads only.
// ---------------------------------------------------------------------------

typedef __bf16 bf16;
typedef __attribute__((ext_vector_type(16))) __bf16 v16bf;
typedef __attribute__((ext_vector_type(8)))  __bf16 v8bf;
typedef __attribute__((ext_vector_type(8)))  float  v8f;

#define DEV static __device__ __forceinline__

// --- A fragment: 16x32 bf16 from row-major source (lda elements/row).
//     Per ISA 7.12.2: lane = m + 16*half; elements 0..7 -> k = e + 8*half,
//     elements 8..15 -> k = 16 + (e-8) + 8*half.  Both halves contiguous.
DEV v16bf frag_a(const bf16* __restrict__ A, int lda) {
  int lane = threadIdx.x & 31;
  int half = lane >> 4, row = lane & 15;
  const bf16* p = A + (size_t)row * lda + half * 8;
  v8bf lo = *(const v8bf*)(p);       // k = 0..7  (+8*half)
  v8bf hi = *(const v8bf*)(p + 16);  // k = 16..23 (+8*half)
  v16bf f;
#pragma unroll
  for (int j = 0; j < 8; ++j) { f[j] = lo[j]; f[8 + j] = hi[j]; }
  return f;
}

// --- B fragment from packed layout.  pk[((kt*NT + nt)*32 + lane)*8 + r]
//     kt = 16-row K subtile index; a 32-K fragment = subtiles (2ks, 2ks+1).
DEV v16bf frag_bp(const bf16* __restrict__ pk, size_t subtile_stride) {
  v8bf lo = *(const v8bf*)(pk);
  v8bf hi = *(const v8bf*)(pk + subtile_stride);
  v16bf f;
#pragma unroll
  for (int j = 0; j < 8; ++j) { f[j] = lo[j]; f[8 + j] = hi[j]; }
  return f;
}

DEV v8f wmma_bf16(v16bf a, v16bf b, v8f c) {
  return __builtin_amdgcn_wmma_f32_16x16x32_bf16(false, a, false, b, (short)0, c,
                                                 false, false);
}

DEV float wave_sum(float v) {
#pragma unroll
  for (int m = 16; m >= 1; m >>= 1) v += __shfl_xor(v, m, 32);
  return v;
}
DEV float wave_max(float v) {
#pragma unroll
  for (int m = 16; m >= 1; m >>= 1) v = fmaxf(v, __shfl_xor(v, m, 32));
  return v;
}

// --- constants --------------------------------------------------------------
#define Bc   8
#define Lc   96
#define Nc   2048
#define Hc   128
#define Ec   16
#define NLc  3
#define Kag  64
#define OUTc 48
#define ROWS (Bc * Nc)  // 16384

// ---------------------------------------------------------------------------
// 1) preprocess
// ---------------------------------------------------------------------------
__global__ void k_preprocess(const float* __restrict__ hist,
                             float* __restrict__ raw, bf16* __restrict__ rawbf,
                             float* __restrict__ meanv, float* __restrict__ stdv,
                             int* __restrict__ dayi, int* __restrict__ weeki) {
  int wave = threadIdx.x >> 5, lane = threadIdx.x & 31;
  int row = blockIdx.x * 8 + wave;
  int b = row >> 11, n = row & (Nc - 1);
  float vals[3];
#pragma unroll
  for (int j = 0; j < 3; ++j) {
    int l = lane + 32 * j;
    vals[j] = hist[(((size_t)(b * Lc + l)) * Nc + n) * 3 + 0];
  }
  float s  = vals[0] + vals[1] + vals[2];
  float sq = vals[0] * vals[0] + vals[1] * vals[1] + vals[2] * vals[2];
  s  = wave_sum(s);
  sq = wave_sum(sq);
  float mean = s / 96.0f;
  float var  = (sq - 96.0f * mean * mean) / 95.0f;  // unbiased
  float sd   = sqrtf(var + 1e-5f);
#pragma unroll
  for (int j = 0; j < 3; ++j) {
    int l = lane + 32 * j;
    float r = (vals[j] - mean) / sd;
    raw[(size_t)row * Lc + l]   = r;
    rawbf[(size_t)row * Lc + l] = (bf16)r;
  }
  if (lane == 0) {
    meanv[row] = mean;
    stdv[row]  = sd;
    float dv = hist[(((size_t)(b * Lc + (Lc - 1))) * Nc + n) * 3 + 1];
    float wv = hist[(((size_t)(b * Lc + (Lc - 1))) * Nc + n) * 3 + 2];
    int di = (int)(dv * 96.0f); di = di < 0 ? 0 : (di > 95 ? 95 : di);
    int wi = (int)(wv * 7.0f);  wi = wi < 0 ? 0 : (wi > 6  ? 6  : wi);
    dayi[row]  = di;
    weeki[row] = wi;
  }
}

// ---------------------------------------------------------------------------
// 2) all 4 routers (input + NL layers)
// ---------------------------------------------------------------------------
__global__ void k_router(const float* __restrict__ raw,
                         const int* __restrict__ dayi, const int* __restrict__ weeki,
                         const float* W0, const float* b0, const float* d0,
                         const float* w0, const float* n0,
                         const float* Wl, const float* bl, const float* dl,
                         const float* wl, const float* nl,
                         float* __restrict__ routers) {
  int wave = threadIdx.x >> 5, lane = threadIdx.x & 31;
  int wg = blockIdx.x * 8 + wave;  // ROWS*4
  int row = wg >> 2, set = wg & 3;
  int n = row & (Nc - 1);
  const float *W, *bb, *dt, *wt, *nt;
  if (set == 0) { W = W0; bb = b0; dt = d0; wt = w0; nt = n0; }
  else {
    int i = set - 1;
    W  = Wl + (size_t)i * Lc * Ec;
    bb = bl + (size_t)i * Ec;
    dt = dl + (size_t)i * 96 * Ec;
    wt = wl + (size_t)i * 7 * Ec;
    nt = nl + (size_t)i * Nc * Ec;
  }
  int e = lane & 15;
  float logit = bb[e] + dt[dayi[row] * Ec + e] + wt[weeki[row] * Ec + e] +
                nt[n * Ec + e];
  const float* rr = raw + (size_t)row * Lc;
  for (int l = 0; l < Lc; ++l) logit += rr[l] * W[l * Ec + e];
  float mx = logit;
#pragma unroll
  for (int m = 8; m >= 1; m >>= 1) mx = fmaxf(mx, __shfl_xor(mx, m, 32));
  float ex = __expf(logit - mx);
  float sm = ex;
#pragma unroll
  for (int m = 8; m >= 1; m >>= 1) sm += __shfl_xor(sm, m, 32);
  if (lane < 16)
    routers[((size_t)set * ROWS + row) * Ec + e] = ex / sm;
}

// ---------------------------------------------------------------------------
// 3) conversions / packing
// ---------------------------------------------------------------------------
__global__ void k_cvt_bf16(const float* __restrict__ in, bf16* __restrict__ out,
                           int nelem) {
  int i = blockIdx.x * 256 + threadIdx.x;
  if (i < nelem) out[i] = (bf16)in[i];
}

// pack f32 row-major (Kd x Nd) weight into fragment-native bf16 layout
__global__ void k_pack_w(const float* __restrict__ src, bf16* __restrict__ dst,
                         int Kd, int Nd) {
  int idx = blockIdx.x * 256 + threadIdx.x;
  if (idx >= Kd * Nd) return;
  int k = idx / Nd, n = idx % Nd;
  int kt = k >> 4, kl = k & 15;
  int lane = (n & 15) + ((kl >> 3) << 4);
  dst[(((size_t)kt * (Nd >> 4) + (n >> 4)) * 32 + lane) * 8 + (kl & 7)] =
      (bf16)src[idx];
}

// pack bf16 row-major (Kd x Nd, leading dim ld) into fragment-native layout
__global__ void k_pack_b(const bf16* __restrict__ src, int ld,
                         bf16* __restrict__ dst, int Kd, int Nd) {
  int idx = blockIdx.x * 256 + threadIdx.x;
  if (idx >= Kd * Nd) return;
  int k = idx / Nd, n = idx % Nd;
  int kt = k >> 4, kl = k & 15;
  int lane = (n & 15) + ((kl >> 3) << 4);
  dst[(((size_t)kt * (Nd >> 4) + (n >> 4)) * 32 + lane) * 8 + (kl & 7)] =
      src[(size_t)k * ld + n];
}

// ---------------------------------------------------------------------------
// 4) Fused GLU + router-weighted expert reduction (WMMA, packed weights).
// ---------------------------------------------------------------------------
template <int KD>
__global__ void k_glu(const bf16* __restrict__ A, const bf16* __restrict__ Wp,
                      const float* __restrict__ bias,
                      const float* __restrict__ router,
                      float* __restrict__ outS) {
  constexpr int NK = KD / 32;
  constexpr size_t SUB = (size_t)256 * 32 * 8;  // packed elems per 16-K subtile
  int wave = threadIdx.x >> 5, lane = threadIdx.x & 31;
  int half = lane >> 4, ncol = lane & 15;
  int mbase = blockIdx.x * 16;
  int h0 = wave * 16;
  int nt0 = h0 >> 4;

  v16bf afr[NK];
#pragma unroll
  for (int ks = 0; ks < NK; ++ks)
    afr[ks] = frag_a(A + (size_t)mbase * KD + ks * 32, KD);

  float xr[8] = {0.f, 0.f, 0.f, 0.f, 0.f, 0.f, 0.f, 0.f};
  for (int e = 0; e < Ec; ++e) {
    int ntA = e * 8 + nt0;        // a-part column tile
    int ntG = 128 + e * 8 + nt0;  // gate column tile
    v8f accA = {0.f, 0.f, 0.f, 0.f, 0.f, 0.f, 0.f, 0.f};
    v8f accG = {0.f, 0.f, 0.f, 0.f, 0.f, 0.f, 0.f, 0.f};
#pragma unroll
    for (int ks = 0; ks < NK; ++ks) {
      const bf16* pb = Wp + ((size_t)(2 * ks) * 256 + ntA) * 256 + lane * 8;
      accA = wmma_bf16(afr[ks], frag_bp(pb, SUB), accA);
    }
#pragma unroll
    for (int ks = 0; ks < NK; ++ks) {
      const bf16* pb = Wp + ((size_t)(2 * ks) * 256 + ntG) * 256 + lane * 8;
      accG = wmma_bf16(afr[ks], frag_bp(pb, SUB), accG);
    }
    float ba = bias[e * Hc + h0 + ncol];
    float bg = bias[2048 + e * Hc + h0 + ncol];
#pragma unroll
    for (int r = 0; r < 8; ++r) {
      int m = mbase + r + 8 * half;
      float av = accA[r] + ba;
      float gv = accG[r] + bg;
      float rv = router[(size_t)m * Ec + e];
      xr[r] += rv * av * (1.f / (1.f + __expf(-gv)));
    }
  }
#pragma unroll
  for (int r = 0; r < 8; ++r)
    outS[(size_t)(mbase + r + 8 * half) * Hc + h0 + ncol] = xr[r];
}

// ---------------------------------------------------------------------------
// 5) Generic batched WMMA GEMM, packed B.  Optional f32 / row-major bf16 /
//    packed bf16 outputs; bias; relu.  Block = 4 waves, one wave per N tile.
// ---------------------------------------------------------------------------
__global__ void k_gemm(const bf16* __restrict__ A, int lda, long long sA,
                       const bf16* __restrict__ Bp, long long sB,
                       float* __restrict__ Cf, bf16* __restrict__ Cbf,
                       int ldc, long long sC, bf16* __restrict__ Cpk,
                       const float* __restrict__ bias, int relu,
                       int Nd, int Kd) {
  int wave = threadIdx.x >> 5, lane = threadIdx.x & 31;
  int half = lane >> 4, ncol = lane & 15;
  int NT = Nd >> 4;
  int ntile = blockIdx.y * 4 + wave;
  if (ntile >= NT) return;
  int mtile = blockIdx.x;
  int mbase = mtile * 16;
  int bt = blockIdx.z;
  A  += (size_t)bt * sA;
  Bp += (size_t)bt * sB;
  size_t sub = (size_t)NT * 256;  // packed elems per 16-K subtile

  v8f acc = {0.f, 0.f, 0.f, 0.f, 0.f, 0.f, 0.f, 0.f};
  for (int kk = 0; kk < Kd; kk += 32) {
    v16bf a = frag_a(A + (size_t)mbase * lda + kk, lda);
    const bf16* pb = Bp + ((size_t)(kk >> 4) * NT + ntile) * 256 + lane * 8;
    acc = wmma_bf16(a, frag_bp(pb, sub), acc);
  }
  int n = ntile * 16 + ncol;
  float bv = bias ? bias[n] : 0.f;
  float vout[8];
#pragma unroll
  for (int r = 0; r < 8; ++r) {
    float v = acc[r] + bv;
    if (relu) v = fmaxf(v, 0.f);
    vout[r] = v;
  }
  if (Cf) {
#pragma unroll
    for (int r = 0; r < 8; ++r)
      Cf[(size_t)bt * sC + (size_t)(mbase + r + 8 * half) * ldc + n] = vout[r];
  }
  if (Cbf) {
#pragma unroll
    for (int r = 0; r < 8; ++r)
      Cbf[(size_t)bt * sC + (size_t)(mbase + r + 8 * half) * ldc + n] =
          (bf16)vout[r];
  }
  if (Cpk) {  // accumulator layout == packed-B layout (k <- m)
    bf16* pd = Cpk + (((size_t)mtile * NT + ntile) * 32 + lane) * 8;
#pragma unroll
    for (int r = 0; r < 8; ++r) pd[r] = (bf16)vout[r];
  }
}

// ---------------------------------------------------------------------------
// 6) embeddings add: x = s + tod + dow + node; skip[0] into h_bf
// ---------------------------------------------------------------------------
__global__ void k_embed(const float* __restrict__ s, const int* __restrict__ dayi,
                        const int* __restrict__ weeki,
                        const float* __restrict__ tod, const float* __restrict__ dow,
                        const float* __restrict__ node,
                        float* __restrict__ x, bf16* __restrict__ xbf,
                        bf16* __restrict__ hbf) {
  int idx = blockIdx.x * 256 + threadIdx.x;  // ROWS*128
  int row = idx >> 7, h = idx & 127;
  int n = row & (Nc - 1);
  float v = s[idx] + tod[dayi[row] * Hc + h] + dow[weeki[row] * Hc + h] +
            node[n * Hc + h];
  x[idx]   = v;
  xbf[idx] = (bf16)v;
  hbf[(size_t)row * 512 + h] = (bf16)v;
}

// ---------------------------------------------------------------------------
// 7) agent proj: split + transpose + pack qa^T / ka^T (128x64 each)
// ---------------------------------------------------------------------------
__global__ void k_split_agent(const bf16* __restrict__ ag, bf16* __restrict__ qaTp,
                              bf16* __restrict__ kaTp) {
  int idx = blockIdx.x * 256 + threadIdx.x;  // 128*64
  int d = idx >> 6, kk = idx & 63;
  int kt = d >> 4, kl = d & 15;
  int lane = (kk & 15) + ((kl >> 3) << 4);
  size_t o = (((size_t)kt * 4 + (kk >> 4)) * 32 + lane) * 8 + (kl & 7);
  qaTp[o] = ag[(size_t)kk * 256 + d];
  kaTp[o] = ag[(size_t)kk * 256 + Hc + d];
}

// ---------------------------------------------------------------------------
// 8) softmax over n (agent attn): S [8][2048][64] -> P [8][64][2048] bf16
// ---------------------------------------------------------------------------
__global__ void k_softmax_agent(const float* __restrict__ S, bf16* __restrict__ P,
                                float scale) {
  int wave = threadIdx.x >> 5, lane = threadIdx.x & 31;
  int wg = blockIdx.x * 8 + wave;  // 512
  int b = wg >> 6, kk = wg & 63;
  float v[64];
  float mx = -1e30f;
#pragma unroll
  for (int i = 0; i < 64; ++i) {
    int n = lane + 32 * i;
    v[i] = S[((size_t)b * Nc + n) * 64 + kk] * scale;
    mx = fmaxf(mx, v[i]);
  }
  mx = wave_max(mx);
  float sm = 0.f;
#pragma unroll
  for (int i = 0; i < 64; ++i) { v[i] = __expf(v[i] - mx); sm += v[i]; }
  sm = wave_sum(sm);
  float inv = 1.f / sm;
#pragma unroll
  for (int i = 0; i < 64; ++i) {
    int n = lane + 32 * i;
    P[((size_t)b * 64 + kk) * Nc + n] = (bf16)(v[i] * inv);
  }
}

// 9) softmax over 64 agents per row
__global__ void k_softmax_rows64(const float* __restrict__ S, bf16* __restrict__ P,
                                 float scale) {
  int wave = threadIdx.x >> 5, lane = threadIdx.x & 31;
  int row = blockIdx.x * 8 + wave;
  float a = S[(size_t)row * 64 + lane] * scale;
  float b = S[(size_t)row * 64 + 32 + lane] * scale;
  float mx = wave_max(fmaxf(a, b));
  a = __expf(a - mx);
  b = __expf(b - mx);
  float sm = wave_sum(a + b);
  P[(size_t)row * 64 + lane]      = (bf16)(a / sm);
  P[(size_t)row * 64 + 32 + lane] = (bf16)(b / sm);
}

// ---------------------------------------------------------------------------
// 10) x = rmsnorm(t + x, scale); optionally skip s=t -> h_bf at hoff
// ---------------------------------------------------------------------------
__global__ void k_rmsnorm(const float* __restrict__ t, const float* __restrict__ scale,
                          float* __restrict__ x, bf16* __restrict__ xbf,
                          bf16* __restrict__ hbf, int hoff) {
  int wave = threadIdx.x >> 5, lane = threadIdx.x & 31;
  int row = blockIdx.x * 8 + wave;
  float y[4];
  float sq = 0.f;
#pragma unroll
  for (int j = 0; j < 4; ++j) {
    int h = lane + 32 * j;
    float yv = t[(size_t)row * Hc + h] + x[(size_t)row * Hc + h];
    y[j] = yv;
    sq += yv * yv;
  }
  sq = wave_sum(sq);
  float inv = 1.f / (sqrtf(sq / 128.f) + 1e-8f);
#pragma unroll
  for (int j = 0; j < 4; ++j) {
    int h = lane + 32 * j;
    float o = scale[h] * y[j] * inv;
    x[(size_t)row * Hc + h]   = o;
    xbf[(size_t)row * Hc + h] = (bf16)o;
    if (hbf) hbf[(size_t)row * 512 + hoff + h] = (bf16)t[(size_t)row * Hc + h];
  }
}

// ---------------------------------------------------------------------------
// 11) de-normalize + transpose to (B, OUT, N, 1)
// ---------------------------------------------------------------------------
__global__ void k_final(const float* __restrict__ o, const float* __restrict__ meanv,
                        const float* __restrict__ stdv, float* __restrict__ out) {
  int idx = blockIdx.x * 256 + threadIdx.x;
  if (idx >= ROWS * OUTc) return;
  int row = idx / OUTc, t = idx % OUTc;
  int b = row >> 11, n = row & (Nc - 1);
  out[((size_t)(b * OUTc + t)) * Nc + n] = o[(size_t)row * OUTc + t] * stdv[row] +
                                           meanv[row];
}

// ---------------------------------------------------------------------------
// Host orchestration
// ---------------------------------------------------------------------------
extern "C" void kernel_launch(void* const* d_in, const int* in_sizes, int n_in,
                              void* d_out, int out_size, void* d_ws, size_t ws_size,
                              hipStream_t stream) {
  const float* hist        = (const float*)d_in[0];
  const float* agent       = (const float*)d_in[1];
  const float* in_router_W = (const float*)d_in[2];
  const float* in_router_b = (const float*)d_in[3];
  const float* in_router_d = (const float*)d_in[4];
  const float* in_router_w = (const float*)d_in[5];
  const float* in_router_n = (const float*)d_in[6];
  const float* in_glu_W    = (const float*)d_in[7];
  const float* in_glu_b    = (const float*)d_in[8];
  const float* tod_emb     = (const float*)d_in[9];
  const float* dow_emb     = (const float*)d_in[10];
  const float* node_emb    = (const float*)d_in[11];
  const float* qkv_W       = (const float*)d_in[12];
  const float* qkv_b       = (const float*)d_in[13];
  const float* agent_W     = (const float*)d_in[14];
  const float* agent_b     = (const float*)d_in[15];
  const float* fc1_W       = (const float*)d_in[16];
  const float* fc1_b       = (const float*)d_in[17];
  const float* fc2_W       = (const float*)d_in[18];
  const float* fc2_b       = (const float*)d_in[19];
  const float* aaga_scale  = (const float*)d_in[20];
  const float* rt_W        = (const float*)d_in[21];
  const float* rt_b        = (const float*)d_in[22];
  const float* rt_d        = (const float*)d_in[23];
  const float* rt_w        = (const float*)d_in[24];
  const float* rt_n        = (const float*)d_in[25];
  const float* moe_W       = (const float*)d_in[26];
  const float* moe_b       = (const float*)d_in[27];
  const float* moe_scale   = (const float*)d_in[28];
  const float* out_W1      = (const float*)d_in[29];
  const float* out_b1      = (const float*)d_in[30];
  const float* out_W2      = (const float*)d_in[31];
  const float* out_b2      = (const float*)d_in[32];
  float* out = (float*)d_out;

  // --- workspace carve-out --------------------------------------------------
  char* ws = (char*)d_ws;
  size_t off = 0;
  auto carve = [&](size_t bytes) -> void* {
    void* p = ws + off;
    off += (bytes + 255) & ~(size_t)255;
    return p;
  };
  float* raw_f   = (float*)carve((size_t)ROWS * Lc * 4);
  bf16*  raw_bf  = (bf16*)carve((size_t)ROWS * Lc * 2);
  float* meanv   = (float*)carve((size_t)ROWS * 4);
  float* stdv    = (float*)carve((size_t)ROWS * 4);
  int*   dayi    = (int*)carve((size_t)ROWS * 4);
  int*   weeki   = (int*)carve((size_t)ROWS * 4);
  float* routers = (float*)carve((size_t)4 * ROWS * Ec * 4);
  float* x_f     = (float*)carve((size_t)ROWS * Hc * 4);
  bf16*  x_bf    = (bf16*)carve((size_t)ROWS * Hc * 2);
  float* tmp_f   = (float*)carve((size_t)ROWS * Hc * 4);   // s / fc2-out
  bf16*  h_bf    = (bf16*)carve((size_t)ROWS * 512 * 2);   // skip concat
  bf16*  qkv_bf  = (bf16*)carve((size_t)ROWS * 384 * 2);
  bf16*  v_pk    = (bf16*)carve((size_t)ROWS * Hc * 2);    // packed v
  bf16*  ag_bf   = (bf16*)carve((size_t)Kag * 256 * 2);
  bf16*  qaT_pk  = (bf16*)carve((size_t)Hc * Kag * 2);
  bf16*  kaT_pk  = (bf16*)carve((size_t)Hc * Kag * 2);
  float* S_f     = (float*)carve((size_t)ROWS * 64 * 4);   // scores; reused as o_f
  bf16*  P_bf    = (bf16*)carve((size_t)ROWS * 64 * 2);    // probs (both stages)
  bf16*  va_bf   = (bf16*)carve((size_t)Bc * Kag * Hc * 2);
  bf16*  va2_pk  = (bf16*)carve((size_t)Bc * Kag * Hc * 2);  // packed
  bf16*  vb_bf   = (bf16*)carve((size_t)ROWS * Hc * 2);
  bf16*  h1_bf   = (bf16*)carve((size_t)ROWS * 512 * 2);
  // packed bf16 weights
  bf16* ingluW_pk = (bf16*)carve((size_t)Lc * 4096 * 2);
  bf16* moeW_pk   = (bf16*)carve((size_t)NLc * Hc * 4096 * 2);
  bf16* qkvW_pk   = (bf16*)carve((size_t)NLc * Hc * 384 * 2);
  bf16* agent_bf  = (bf16*)carve((size_t)Kag * Hc * 2);
  bf16* agW_pk    = (bf16*)carve((size_t)NLc * Hc * 256 * 2);
  bf16* fc1W_pk   = (bf16*)carve((size_t)NLc * Hc * Hc * 2);
  bf16* fc2W_pk   = (bf16*)carve((size_t)NLc * Hc * Hc * 2);
  bf16* W1_pk     = (bf16*)carve((size_t)512 * 512 * 2);
  bf16* W2_pk     = (bf16*)carve((size_t)512 * OUTc * 2);
  (void)ws_size; (void)in_sizes; (void)n_in; (void)out_size;

  const float scaleH = 1.0f / sqrtf((float)Hc);

  // 1) preprocess + routers
  k_preprocess<<<ROWS / 8, 256, 0, stream>>>(hist, raw_f, raw_bf, meanv, stdv,
                                             dayi, weeki);
  k_router<<<ROWS * 4 / 8, 256, 0, stream>>>(
      raw_f, dayi, weeki, in_router_W, in_router_b, in_router_d, in_router_w,
      in_router_n, rt_W, rt_b, rt_d, rt_w, rt_n, routers);

  // 2) weight convert+pack
  auto pkw = [&](const float* src, bf16* dst, int Kd, int Nd) {
    k_pack_w<<<(Kd * Nd + 255) / 256, 256, 0, stream>>>(src, dst, Kd, Nd);
  };
  pkw(in_glu_W, ingluW_pk, Lc, 4096);
  for (int i = 0; i < NLc; ++i) {
    pkw(moe_W + (size_t)i * Hc * 4096, moeW_pk + (size_t)i * Hc * 4096, Hc, 4096);
    pkw(qkv_W + (size_t)i * Hc * 384, qkvW_pk + (size_t)i * Hc * 384, Hc, 384);
    pkw(agent_W + (size_t)i * Hc * 256, agW_pk + (size_t)i * Hc * 256, Hc, 256);
    pkw(fc1_W + (size_t)i * Hc * Hc, fc1W_pk + (size_t)i * Hc * Hc, Hc, Hc);
    pkw(fc2_W + (size_t)i * Hc * Hc, fc2W_pk + (size_t)i * Hc * Hc, Hc, Hc);
  }
  pkw(out_W1, W1_pk, 512, 512);
  pkw(out_W2, W2_pk, 512, OUTc);
  k_cvt_bf16<<<(Kag * Hc + 255) / 256, 256, 0, stream>>>(agent, agent_bf,
                                                         Kag * Hc);

  // 3) input GLU (K=96) -> tmp; embeddings -> x (and skip[0])
  k_glu<96><<<ROWS / 16, 256, 0, stream>>>(raw_bf, ingluW_pk, in_glu_b, routers,
                                           tmp_f);
  k_embed<<<ROWS * Hc / 256, 256, 0, stream>>>(tmp_f, dayi, weeki, tod_emb,
                                               dow_emb, node_emb, x_f, x_bf, h_bf);

  for (int i = 0; i < NLc; ++i) {
    // qkv = x @ Wqkv + b  (row-major bf16 out; q/k/v at col offsets, lda=384)
    k_gemm<<<dim3(ROWS / 16, 6, 1), 128, 0, stream>>>(
        x_bf, Hc, 0, qkvW_pk + (size_t)i * Hc * 384, 0, nullptr, qkv_bf, 384, 0,
        nullptr, qkv_b + i * 384, 0, 384, Hc);
    // pack v for the attn@V GEMM (16384x128, ld=384)
    k_pack_b<<<(ROWS * Hc + 255) / 256, 256, 0, stream>>>(qkv_bf + 256, 384,
                                                          v_pk, ROWS, Hc);
    // agent proj (64x256) + split/transpose/pack qa^T, ka^T
    k_gemm<<<dim3(4, 4, 1), 128, 0, stream>>>(
        agent_bf, Hc, 0, agW_pk + (size_t)i * Hc * 256, 0, nullptr, ag_bf, 256,
        0, nullptr, agent_b + i * 256, 0, 256, Hc);
    k_split_agent<<<Hc * Kag / 256, 256, 0, stream>>>(ag_bf, qaT_pk, kaT_pk);
    // S1^T[b,n,k] = k @ qa^T  (batched over b)
    k_gemm<<<dim3(Nc / 16, 1, Bc), 128, 0, stream>>>(
        qkv_bf + Hc, 384, (long long)Nc * 384, qaT_pk, 0, S_f, nullptr, 64,
        (long long)Nc * 64, nullptr, nullptr, 0, 64, Hc);
    k_softmax_agent<<<64, 256, 0, stream>>>(S_f, P_bf, scaleH);
    // va = P1 @ v  (batched; packed v, per-batch stride = 2048*128)
    k_gemm<<<dim3(Kag / 16, 2, Bc), 128, 0, stream>>>(
        P_bf, Nc, (long long)Kag * Nc, v_pk, (long long)Nc * Hc, nullptr, va_bf,
        Hc, (long long)Kag * Hc, nullptr, nullptr, 0, Hc, Nc);
    // va2 = va @ fc1 + b  -> packed output (B operand of vb GEMM)
    k_gemm<<<dim3(Bc * Kag / 16, 2, 1), 128, 0, stream>>>(
        va_bf, Hc, 0, fc1W_pk + (size_t)i * Hc * Hc, 0, nullptr, nullptr, 0, 0,
        va2_pk, fc1_b + i * Hc, 0, Hc, Hc);
    // S2[b,n,k] = q @ ka^T
    k_gemm<<<dim3(ROWS / 16, 1, 1), 128, 0, stream>>>(
        qkv_bf, 384, 0, kaT_pk, 0, S_f, nullptr, 64, 0, nullptr, nullptr, 0, 64,
        Hc);
    k_softmax_rows64<<<ROWS / 8, 256, 0, stream>>>(S_f, P_bf, scaleH);
    // vb = P2 @ va2  (batched; packed va2, per-batch = 4 subtiles * 8NT * 256)
    k_gemm<<<dim3(Nc / 16, 2, Bc), 128, 0, stream>>>(
        P_bf, 64, (long long)Nc * 64, va2_pk, (long long)Kag * Hc, nullptr,
        vb_bf, Hc, (long long)Nc * Hc, nullptr, nullptr, 0, Hc, Kag);
    // t = vb @ fc2 + b
    k_gemm<<<dim3(ROWS / 16, 2, 1), 128, 0, stream>>>(
        vb_bf, Hc, 0, fc2W_pk + (size_t)i * Hc * Hc, 0, tmp_f, nullptr, Hc, 0,
        nullptr, fc2_b + i * Hc, 0, Hc, Hc);
    // x = rmsnorm(t + x)
    k_rmsnorm<<<ROWS / 8, 256, 0, stream>>>(tmp_f, aaga_scale + i * Hc, x_f,
                                            x_bf, nullptr, 0);
    // MoE GLU (K=128) -> s
    k_glu<128><<<ROWS / 16, 256, 0, stream>>>(
        x_bf, moeW_pk + (size_t)i * Hc * 4096, moe_b + (size_t)i * 4096,
        routers + (size_t)(i + 1) * ROWS * Ec, tmp_f);
    // x = rmsnorm(s + x); skip[i+1] = s
    k_rmsnorm<<<ROWS / 8, 256, 0, stream>>>(tmp_f, moe_scale + i * Hc, x_f,
                                            x_bf, h_bf, (i + 1) * Hc);
  }

  // output MLP
  k_gemm<<<dim3(ROWS / 16, 8, 1), 128, 0, stream>>>(
      h_bf, 512, 0, W1_pk, 0, nullptr, h1_bf, 512, 0, nullptr, out_b1, 1, 512,
      512);
  k_gemm<<<dim3(ROWS / 16, 1, 1), 128, 0, stream>>>(
      h1_bf, 512, 0, W2_pk, 0, S_f, nullptr, OUTc, 0, nullptr, out_b2, 0, OUTc,
      512);
  k_final<<<(ROWS * OUTc + 255) / 256, 256, 0, stream>>>(S_f, meanv, stdv, out);
}
